// RNNCellModel_11991548690507
// MI455X (gfx1250) — matrix operand order repaired and checked by reading, not compiled, not run on hardware
//
#include <hip/hip_runtime.h>
#include <hip/hip_bf16.h>

typedef __attribute__((ext_vector_type(16))) __bf16 v16bf;
typedef __attribute__((ext_vector_type(8)))  float  v8f;

union AFrag { v16bf v; int4 q[2]; };

#define B_DIM   64
#define T_DIM   512
#define E_DIM   512
#define H_DIM   1024
#define K_DIM   2048   // 2E + H
#define G_DIM   4096   // 4H
#define V0_DIM  2048
#define V1_DIM  512
#define NV_DIM  2560   // V0+V1

__device__ __forceinline__ unsigned short f2bf(float x) {
    union { float f; unsigned u; } v; v.f = x;
    unsigned r = v.u + 0x7FFFu + ((v.u >> 16) & 1u);   // round-to-nearest-even
    return (unsigned short)(r >> 16);
}

// async global -> LDS copy of 32 bytes (two B128 DMAs, ASYNCcnt-tracked).
// INST_OFFSET is added to BOTH the LDS and global address per CDNA5 ISA §08.
__device__ __forceinline__ void async_copy32(void* lds_dst, const void* gsrc) {
    unsigned lo = (unsigned)(size_t)lds_dst;   // low 32 bits of generic = LDS offset
    asm volatile(
        "global_load_async_to_lds_b128 %0, %1, off\n\t"
        "global_load_async_to_lds_b128 %0, %1, off offset:16"
        :: "v"(lo), "v"(gsrc) : "memory");
}
__device__ __forceinline__ void async_wait0() {
    asm volatile("s_wait_asynccnt 0" ::: "memory");
}
__device__ __forceinline__ void async_wait2() {
    asm volatile("s_wait_asynccnt 2" ::: "memory");
}

// ---------------- weight / bias packing ----------------
__global__ void pack_w_kernel(const float* __restrict__ w_ih,
                              const float* __restrict__ w_hh,
                              unsigned short* __restrict__ Wc) {
    int idx = blockIdx.x * blockDim.x + threadIdx.x;            // [4096 x 2048]
    if (idx >= G_DIM * K_DIM) return;
    int r = idx / K_DIM, c = idx % K_DIM;
    float v = (c < 1024) ? w_ih[r * 1024 + c] : w_hh[r * 1024 + (c - 1024)];
    Wc[idx] = f2bf(v);
}

__global__ void pack_bias_kernel(const float* __restrict__ b_ih,
                                 const float* __restrict__ b_hh,
                                 float* __restrict__ biasc) {
    int i = blockIdx.x * blockDim.x + threadIdx.x;
    if (i < G_DIM) biasc[i] = b_ih[i] + b_hh[i];
}

__global__ void pack_dec_kernel(const float* __restrict__ dec0_w,
                                const float* __restrict__ dec1_w,
                                const float* __restrict__ dec0_b,
                                const float* __restrict__ dec1_b,
                                unsigned short* __restrict__ decW,
                                float* __restrict__ decB) {
    int idx = blockIdx.x * blockDim.x + threadIdx.x;            // [2560 x 1024]
    if (idx < NV_DIM * H_DIM) {
        int r = idx / H_DIM, c = idx % H_DIM;
        float v = (r < V0_DIM) ? dec0_w[r * H_DIM + c] : dec1_w[(r - V0_DIM) * H_DIM + c];
        decW[idx] = f2bf(v);
    }
    if (idx < NV_DIM) decB[idx] = (idx < V0_DIM) ? dec0_b[idx] : dec1_b[idx - V0_DIM];
}

// ---------------- embedding gather -> bf16 [T*B, 2E] ----------------
__global__ void embed_kernel(const long long* __restrict__ in0,
                             const long long* __restrict__ in1,
                             const float* __restrict__ emb0_w,
                             const float* __restrict__ emb1_w,
                             unsigned short* __restrict__ Xemb) {
    int flat = blockIdx.x;                 // t*64 + b
    int t = flat >> 6, b = flat & 63;
    long long r0 = in0[b * T_DIM + t];
    long long r1 = in1[b * T_DIM + t];
    int e = threadIdx.x;                   // 256 threads, 2 iters over E=512
    unsigned short* dst = Xemb + (size_t)flat * 1024;
    for (int i = 0; i < 2; ++i) {
        int ee = e + i * 256;
        dst[ee]       = f2bf(emb0_w[r0 * E_DIM + ee]);
        dst[512 + ee] = f2bf(emb1_w[r1 * E_DIM + ee]);
    }
}

// ---------------- state init ----------------
__global__ void init_state_kernel(const float* __restrict__ h0,
                                  const float* __restrict__ c0,
                                  unsigned short* __restrict__ hbf0,
                                  float* __restrict__ cbuf,
                                  float* __restrict__ hf32) {
    int i = blockIdx.x * blockDim.x + threadIdx.x;
    if (i < B_DIM * H_DIM) {
        hbf0[i] = f2bf(h0[i]);
        cbuf[i] = c0[i];
        hf32[i] = h0[i];
    }
}

// ---------------- one LSTM step ----------------
// grid: 16 blocks (64 h-cols each), 512 threads = 16 waves.
// wave w: m_tile = w%4 (16 batch rows), j = w/4 (16 h-cols), owns all 4 gates
__global__ __launch_bounds__(512) void lstm_step_kernel(
        const unsigned short* __restrict__ Xemb,   // [T*64, 1024]
        const unsigned short* __restrict__ Wc,     // [4096, 2048]
        const float* __restrict__ biasc,           // [4096]
        const unsigned short* __restrict__ hIn,    // [64,1024] bf16
        unsigned short* __restrict__ hOut,         // [64,1024] bf16
        float* __restrict__ cbuf,                  // [64,1024]
        float* __restrict__ hf32,                  // [64,1024]
        unsigned short* __restrict__ outH,         // [B*T, 1024]
        int t) {
    __shared__ __align__(16) unsigned short a_tile[2][64][128];  // 2 x 16 KB K-chunks

    const int tid  = threadIdx.x;
    const int wave = tid >> 5;
    const int lane = tid & 31;
    const int l16  = lane & 15;
    const int hi   = lane >> 4;
    const int m_tile = wave & 3;
    const int j      = wave >> 2;
    const int c0 = blockIdx.x * 64;                // h-column base of this block

    v8f acc[4];
    #pragma unroll
    for (int g = 0; g < 4; ++g) {
        float bv = biasc[g * H_DIM + c0 + j * 16 + l16];
        #pragma unroll
        for (int r = 0; r < 8; ++r) acc[g][r] = bv;
    }

    const int ldm = tid >> 3;        // 0..63 : A row loaded by this thread
    const int ldc = (tid & 7) * 16;  // 0..112: A col (bf16) base

    const int NC = K_DIM / 128;      // 16 chunks
    auto srcptr = [&](int kk) -> const unsigned short* {
        return (kk < 1024) ? (Xemb + (size_t)(t * 64 + ldm) * 1024 + kk + ldc)
                           : (hIn  + (size_t)ldm * 1024 + (kk - 1024) + ldc);
    };

    // prologue: DMA chunk 0 into buffer 0
    async_copy32(&a_tile[0][ldm][ldc], srcptr(0));

    for (int ci = 0; ci < NC; ++ci) {
        const int cur = ci & 1;
        if (ci + 1 < NC) {                        // prefetch next chunk
            async_copy32(&a_tile[cur ^ 1][ldm][ldc], srcptr((ci + 1) * 128));
            async_wait2();                        // older pair (current chunk) done
        } else {
            async_wait0();
        }
        __syncthreads();

        const int kk = ci * 128;
        #pragma unroll
        for (int k2 = 0; k2 < 128; k2 += 32) {
            AFrag af;
            const unsigned short* ar = &a_tile[cur][m_tile * 16 + l16][k2];
            af.q[0] = *(const int4*)(ar + hi * 16);
            af.q[1] = *(const int4*)(ar + hi * 16 + 8);
            AFrag bfg[4];
            #pragma unroll
            for (int g = 0; g < 4; ++g) {         // issue all 8 B loads up front
                const unsigned short* br =
                    Wc + (size_t)(g * H_DIM + c0 + j * 16 + l16) * K_DIM + kk + k2 + hi * 16;
                bfg[g].q[0] = *(const int4*)(br);
                bfg[g].q[1] = *(const int4*)(br + 8);
            }
            #pragma unroll
            for (int g = 0; g < 4; ++g)
                acc[g] = __builtin_amdgcn_wmma_f32_16x16x32_bf16(
                    false, af.v, false, bfg[g].v, (short)0, acc[g], false, false);
        }
        __syncthreads();
    }

    // activations + state update, entirely in-wave
    const int col = c0 + j * 16 + l16;             // h column
    #pragma unroll
    for (int r = 0; r < 8; ++r) {
        int m = m_tile * 16 + 8 * hi + r;          // batch row
        float iv = 1.0f / (1.0f + __expf(-acc[0][r]));
        float fv = 1.0f / (1.0f + __expf(-acc[1][r]));
        float gv = tanhf(acc[2][r]);
        float ov = 1.0f / (1.0f + __expf(-acc[3][r]));
        int sidx = m * H_DIM + col;
        float cn = fv * cbuf[sidx] + iv * gv;
        float hn = ov * tanhf(cn);
        cbuf[sidx] = cn;
        hf32[sidx] = hn;
        unsigned short hb = f2bf(hn);
        hOut[sidx] = hb;
        outH[(size_t)(m * T_DIM + t) * H_DIM + col] = hb;   // row = b*T + t
    }
}

// ---------------- decoder GEMM: [32768,1024] x [2560,1024]^T + bias ----------------
// grid (256, 20), 256 threads = 8 waves; block tile 128M x 128N
__global__ __launch_bounds__(256) void decoder_kernel(
        const unsigned short* __restrict__ outH,   // [32768, 1024] bf16
        const unsigned short* __restrict__ decW,   // [2560, 1024] bf16
        const float* __restrict__ decB,            // [2560]
        float* __restrict__ out) {                 // [32768, 2560]
    __shared__ __align__(16) unsigned short b_tile[2][128][32];  // 2 x 8 KB K-chunks

    const int tid  = threadIdx.x;
    const int wave = tid >> 5;
    const int lane = tid & 31;
    const int l16  = lane & 15;
    const int hi   = lane >> 4;
    const int m0 = blockIdx.x * 128;
    const int n0 = blockIdx.y * 128;
    const int mrow = m0 + wave * 16 + l16;         // A row for this lane's fragment

    v8f acc[8];
    #pragma unroll
    for (int nt = 0; nt < 8; ++nt) {
        float bv = decB[n0 + nt * 16 + l16];
        #pragma unroll
        for (int r = 0; r < 8; ++r) acc[nt][r] = bv;
    }

    const int ldn = tid >> 1;                      // 0..127
    const int ldc = (tid & 1) * 16;                // 0 / 16

    const int NC = H_DIM / 32;                     // 32 chunks
    async_copy32(&b_tile[0][ldn][ldc], decW + (size_t)(n0 + ldn) * H_DIM + ldc);

    for (int ci = 0; ci < NC; ++ci) {
        const int cur = ci & 1;
        if (ci + 1 < NC) {
            async_copy32(&b_tile[cur ^ 1][ldn][ldc],
                         decW + (size_t)(n0 + ldn) * H_DIM + (ci + 1) * 32 + ldc);
            async_wait2();
        } else {
            async_wait0();
        }
        __syncthreads();

        const int kk = ci * 32;
        AFrag af;
        const unsigned short* ar = outH + (size_t)mrow * H_DIM + kk;
        af.q[0] = *(const int4*)(ar + hi * 16);
        af.q[1] = *(const int4*)(ar + hi * 16 + 8);
        #pragma unroll
        for (int nt = 0; nt < 8; ++nt) {
            AFrag bfg;
            const unsigned short* br = &b_tile[cur][nt * 16 + l16][hi * 16];
            bfg.q[0] = *(const int4*)(br);
            bfg.q[1] = *(const int4*)(br + 8);
            acc[nt] = __builtin_amdgcn_wmma_f32_16x16x32_bf16(
                false, af.v, false, bfg.v, (short)0, acc[nt], false, false);
        }
        __syncthreads();
    }

    #pragma unroll
    for (int nt = 0; nt < 8; ++nt) {
        int col = n0 + nt * 16 + l16;
        #pragma unroll
        for (int r = 0; r < 8; ++r) {
            int row = m0 + wave * 16 + 8 * hi + r;
            out[(size_t)row * NV_DIM + col] = acc[nt][r];
        }
    }
}

// ---------------- tail: h_f, c_f ----------------
__global__ void tail_kernel(const float* __restrict__ hf32,
                            const float* __restrict__ cbuf,
                            float* __restrict__ out_tail) {
    int i = blockIdx.x * blockDim.x + threadIdx.x;
    int n = B_DIM * H_DIM;
    if (i < n)          out_tail[i] = hf32[i];
    else if (i < 2 * n) out_tail[i] = cbuf[i - n];
}

extern "C" void kernel_launch(void* const* d_in, const int* in_sizes, int n_in,
                              void* d_out, int out_size, void* d_ws, size_t ws_size,
                              hipStream_t stream) {
    const long long* in0   = (const long long*)d_in[0];
    const long long* in1   = (const long long*)d_in[1];
    const float* emb0_w    = (const float*)d_in[2];
    const float* emb1_w    = (const float*)d_in[3];
    const float* w_ih      = (const float*)d_in[4];
    const float* b_ih      = (const float*)d_in[5];
    const float* w_hh      = (const float*)d_in[6];
    const float* b_hh      = (const float*)d_in[7];
    const float* dec0_w    = (const float*)d_in[8];
    const float* dec0_b    = (const float*)d_in[9];
    const float* dec1_w    = (const float*)d_in[10];
    const float* dec1_b    = (const float*)d_in[11];
    const float* h0        = (const float*)d_in[12];
    const float* c0        = (const float*)d_in[13];
    float* out = (float*)d_out;

    // workspace carve-up (256B aligned)
    char* ws = (char*)d_ws;
    size_t off = 0;
    auto alloc = [&](size_t bytes) { char* p = ws + off; off = (off + bytes + 255) & ~(size_t)255; return p; };
    unsigned short* Xemb  = (unsigned short*)alloc((size_t)T_DIM * B_DIM * 1024 * sizeof(unsigned short));
    unsigned short* Wc    = (unsigned short*)alloc((size_t)G_DIM * K_DIM * sizeof(unsigned short));
    float*          biasc = (float*)alloc(G_DIM * sizeof(float));
    unsigned short* decW  = (unsigned short*)alloc((size_t)NV_DIM * H_DIM * sizeof(unsigned short));
    float*          decB  = (float*)alloc(NV_DIM * sizeof(float));
    unsigned short* hA    = (unsigned short*)alloc((size_t)B_DIM * H_DIM * sizeof(unsigned short));
    unsigned short* hB    = (unsigned short*)alloc((size_t)B_DIM * H_DIM * sizeof(unsigned short));
    float*          cbuf  = (float*)alloc((size_t)B_DIM * H_DIM * sizeof(float));
    float*          hf32  = (float*)alloc((size_t)B_DIM * H_DIM * sizeof(float));
    unsigned short* outH  = (unsigned short*)alloc((size_t)B_DIM * T_DIM * H_DIM * sizeof(unsigned short));
    (void)ws_size; (void)in_sizes; (void)n_in; (void)out_size;

    // pack weights / biases to bf16
    pack_w_kernel<<<(G_DIM * K_DIM + 255) / 256, 256, 0, stream>>>(w_ih, w_hh, Wc);
    pack_bias_kernel<<<(G_DIM + 255) / 256, 256, 0, stream>>>(b_ih, b_hh, biasc);
    pack_dec_kernel<<<(NV_DIM * H_DIM + 255) / 256, 256, 0, stream>>>(dec0_w, dec1_w, dec0_b, dec1_b, decW, decB);

    // embedding gather
    embed_kernel<<<T_DIM * B_DIM, 256, 0, stream>>>(in0, in1, emb0_w, emb1_w, Xemb);

    // init state
    init_state_kernel<<<(B_DIM * H_DIM + 255) / 256, 256, 0, stream>>>(h0, c0, hA, cbuf, hf32);

    // sequential LSTM: double-buffered h
    for (int t = 0; t < T_DIM; ++t) {
        const unsigned short* hin = (t & 1) ? hB : hA;
        unsigned short* hout      = (t & 1) ? hA : hB;
        lstm_step_kernel<<<H_DIM / 64, 512, 0, stream>>>(
            Xemb, Wc, biasc, hin, hout, cbuf, hf32, outH, t);
    }

    // decoder logits
    dim3 dgrid(B_DIM * T_DIM / 128, NV_DIM / 128);
    decoder_kernel<<<dgrid, 256, 0, stream>>>(outH, decW, decB, out);

    // h_f, c_f tail
    tail_kernel<<<(2 * B_DIM * H_DIM + 255) / 256, 256, 0, stream>>>(
        hf32, cbuf, out + (size_t)B_DIM * T_DIM * NV_DIM);
}
